// GNN_81046032875948
// MI455X (gfx1250) — matrix operand order repaired and checked by reading, not compiled, not run on hardware
//
#include <hip/hip_runtime.h>
#include <hip/hip_bf16.h>

typedef _Float16 f16;
typedef __attribute__((ext_vector_type(16))) _Float16 v16h;
typedef __attribute__((ext_vector_type(8)))  float    v8f;

#define N_NODES 50000
#define N_EDGES 300000
#define HIDDEN  128
#define NA_ST   208   // padded per-node feature row stride (halves), 16B-aligned rows
#define SH_ST   136   // sH row stride (halves): 136 = 8 mod 16 -> conflict-free b128 reads

union H8 { uint4 u; f16 h[8]; };
union H4 { uint2 u; f16 h[4]; };

__device__ __forceinline__ float gelu_f(float x) {
  // exact GELU (approximate=False in reference)
  return 0.5f * x * (1.0f + erff(x * 0.70710678118654752f));
}

union ABFrag { uint4 u[2]; v16h h; };

// One 16(M)x16(N) WMMA tile: A from LDS (f16 row-major, strideA halves),
// B from global f16 weights, row-major [Kdepth x 128]. Returns bias + A*B.
__device__ __forceinline__ v8f wmma_tile(const f16* sA, int strideA, int rowBase,
                                         int Kdepth, const f16* __restrict__ W,
                                         int n0, float bias) {
  const int lane   = threadIdx.x & 31;
  const int laneLo = lane & 15;
  const int hiOff  = (lane & 16) ? 8 : 0;
  v8f acc;
#pragma unroll
  for (int i = 0; i < 8; ++i) acc[i] = bias;
  for (int kc = 0; kc < Kdepth; kc += 32) {
    // A fragment: lanes 0-15 hold K = kc+[0..7,16..23]; lanes 16-31: +8
    const f16* pa = sA + (rowBase + laneLo) * strideA + kc + hiOff;
    ABFrag a, b;
    a.u[0] = *(const uint4*)(pa);
    a.u[1] = *(const uint4*)(pa + 16);
    // B fragment: lane L holds column n0+(L&15), K = kc+[0..7,16..23]; lanes 16-31: +8
    const f16* pb = W + (kc + hiOff) * HIDDEN + n0 + laneLo;
#pragma unroll
    for (int s = 0; s < 8; ++s) {
      b.h[s]     = pb[s * HIDDEN];
      b.h[8 + s] = pb[(16 + s) * HIDDEN];
    }
    acc = __builtin_amdgcn_wmma_f32_16x16x32_f16(false, a.h, false, b.h,
                                                 (short)0, acc, false, false);
  }
  return acc;
}

// ---------------- fused edge MLP: gather -> msg(LDS) -> GEMM1 -> GEMM2 -> atomic agg
// msg layout (section-padded, all section starts 16B aligned):
//   [0,naP): 0.5*(na[s]+na[d]) ; [naP,2naP): 0.5*|na[s]-na[d]| ; [2naP, 2naP+eaPad): ea
__global__ __launch_bounds__(256) void edge_mlp(
    const f16* __restrict__ na, int naP,
    const f16* __restrict__ ea, int eaPad,
    const int* __restrict__ src, const int* __restrict__ dst,
    const f16* __restrict__ W1, const float* __restrict__ b1,
    const f16* __restrict__ W2, const float* __restrict__ b2,
    int Kpad, float* __restrict__ agg)
{
  extern __shared__ f16 smem[];
  const int SA = Kpad + 8;          // 8 mod 16 halves -> conflict-free A reads
  f16* sA = smem;                   // 64 x SA
  f16* sH = smem + 64 * SA;         // 64 x SH_ST
  const int tid = threadIdx.x;
  const int e0  = blockIdx.x * 64;

  { // cooperative message build: 4 threads per edge row, 16B chunks
    const int r  = tid >> 2;
    const int c4 = tid & 3;
    const int e  = e0 + r;
    const bool valid = (e < N_EDGES);
    f16* row = sA + r * SA;
    uint4 zz; zz.x = zz.y = zz.z = zz.w = 0u;
    if (valid) {
      const int s = src[e], d = dst[e];
      const f16* ns = na + (unsigned)s * NA_ST;
      const f16* nd = na + (unsigned)d * NA_ST;
      const int naChunks = naP >> 3;
      for (int ch = c4; ch < naChunks; ch += 4) {
        H8 a, b, oA, oD;
        a.u = *(const uint4*)(ns + ch * 8);
        b.u = *(const uint4*)(nd + ch * 8);
#pragma unroll
        for (int q = 0; q < 8; ++q) {
          const float fa = (float)a.h[q], fb = (float)b.h[q];
          oA.h[q] = (f16)(0.5f * (fa + fb));
          oD.h[q] = (f16)(0.5f * fabsf(fa - fb));
        }
        *(uint4*)(row + ch * 8)       = oA.u;
        *(uint4*)(row + naP + ch * 8) = oD.u;
      }
      const f16* pe = ea + (unsigned)e * HIDDEN;
      for (int ch = c4; ch < (eaPad >> 3); ch += 4)
        *(uint4*)(row + 2 * naP + ch * 8) = *(const uint4*)(pe + ch * 8);
      for (int c0 = 2 * naP + eaPad + c4 * 8; c0 < Kpad; c0 += 32)
        *(uint4*)(row + c0) = zz;
    } else {
      for (int c0 = c4 * 8; c0 < Kpad; c0 += 32) *(uint4*)(row + c0) = zz;
    }
  }
  __syncthreads();

  const int wave  = tid >> 5;
  const int lane  = tid & 31;
  const int n0    = wave * 16;
  const int colLo = lane & 15;
  const int rHi   = (lane & 16) ? 8 : 0;
  const float bias1 = b1[n0 + colLo];
  const float bias2 = b2[n0 + colLo];

#pragma unroll
  for (int mt = 0; mt < 4; ++mt) {
    v8f acc = wmma_tile(sA, SA, mt * 16, Kpad, W1, n0, bias1);
#pragma unroll
    for (int i = 0; i < 8; ++i)
      sH[(mt * 16 + rHi + i) * SH_ST + n0 + colLo] = (f16)gelu_f(acc[i]);
  }
  __syncthreads();

#pragma unroll
  for (int mt = 0; mt < 4; ++mt) {
    v8f acc = wmma_tile(sH, SH_ST, mt * 16, HIDDEN, W2, n0, bias2);
#pragma unroll
    for (int i = 0; i < 8; ++i) {
      const int e = e0 + mt * 16 + rHi + i;
      if (e < N_EDGES) {
        const int node = src[e];
        atomicAdd(&agg[(unsigned)node * HIDDEN + n0 + colLo], gelu_f(acc[i]));
      }
    }
  }
}

// ---------------- fused node MLP: [na | agg/deg](LDS) -> GEMM1 -> GEMM2 -> x
// A layout: [0,naP): na ; [naP, naP+128): agg/deg
__global__ __launch_bounds__(256) void node_mlp(
    const f16* __restrict__ na, int naP,
    const float* __restrict__ agg, const float* __restrict__ deg,
    const f16* __restrict__ W1, const float* __restrict__ b1,
    const f16* __restrict__ W2, const float* __restrict__ b2,
    int Kpad, float* __restrict__ xout)
{
  extern __shared__ f16 smem[];
  const int SA = Kpad + 8;
  f16* sA = smem;
  f16* sH = smem + 64 * SA;
  const int tid = threadIdx.x;
  const int v0  = blockIdx.x * 64;

  {
    const int r  = tid >> 2;
    const int c4 = tid & 3;
    const int nr = v0 + r;
    const bool valid = (nr < N_NODES);
    f16* row = sA + r * SA;
    uint4 zz; zz.x = zz.y = zz.z = zz.w = 0u;
    if (valid) {
      const f16* nrow = na + (unsigned)nr * NA_ST;
      for (int ch = c4; ch < (naP >> 3); ch += 4)
        *(uint4*)(row + ch * 8) = *(const uint4*)(nrow + ch * 8);
      const float inv = 1.0f / fmaxf(deg[nr], 1.0f);
      const float* ag = agg + (unsigned)nr * HIDDEN;
      for (int ch = c4; ch < 16; ch += 4) {   // 128/8 chunks
        const float4 f0 = *(const float4*)(ag + ch * 8);
        const float4 f1 = *(const float4*)(ag + ch * 8 + 4);
        H8 o;
        o.h[0] = (f16)(f0.x * inv); o.h[1] = (f16)(f0.y * inv);
        o.h[2] = (f16)(f0.z * inv); o.h[3] = (f16)(f0.w * inv);
        o.h[4] = (f16)(f1.x * inv); o.h[5] = (f16)(f1.y * inv);
        o.h[6] = (f16)(f1.z * inv); o.h[7] = (f16)(f1.w * inv);
        *(uint4*)(row + naP + ch * 8) = o.u;
      }
      for (int c0 = naP + HIDDEN + c4 * 8; c0 < Kpad; c0 += 32)
        *(uint4*)(row + c0) = zz;
    } else {
      for (int c0 = c4 * 8; c0 < Kpad; c0 += 32) *(uint4*)(row + c0) = zz;
    }
  }
  __syncthreads();

  const int wave  = tid >> 5;
  const int lane  = tid & 31;
  const int n0    = wave * 16;
  const int colLo = lane & 15;
  const int rHi   = (lane & 16) ? 8 : 0;
  const float bias1 = b1[n0 + colLo];
  const float bias2 = b2[n0 + colLo];

#pragma unroll
  for (int mt = 0; mt < 4; ++mt) {
    v8f acc = wmma_tile(sA, SA, mt * 16, Kpad, W1, n0, bias1);
#pragma unroll
    for (int i = 0; i < 8; ++i)
      sH[(mt * 16 + rHi + i) * SH_ST + n0 + colLo] = (f16)gelu_f(acc[i]);
  }
  __syncthreads();

#pragma unroll
  for (int mt = 0; mt < 4; ++mt) {
    v8f acc = wmma_tile(sH, SH_ST, mt * 16, HIDDEN, W2, n0, bias2);
#pragma unroll
    for (int i = 0; i < 8; ++i) {
      const int nr = v0 + mt * 16 + rHi + i;
      if (nr < N_NODES)
        xout[(unsigned)nr * HIDDEN + n0 + colLo] = gelu_f(acc[i]);
    }
  }
}

// ---------------- helper kernels ----------------
__global__ void zero_f32(float* p, unsigned n) {
  for (unsigned i = blockIdx.x * blockDim.x + threadIdx.x; i < n;
       i += gridDim.x * blockDim.x) p[i] = 0.0f;
}

// f32 -> f16 weight conversion with up to 3 row-range remaps (padded<-orig); rest zero.
__global__ void cvt_w16_seg(const float* __restrict__ w, f16* __restrict__ out, int Kpad,
                            int p0, int o0, int l0,
                            int p1, int o1, int l1,
                            int p2, int o2, int l2) {
  const unsigned total = (unsigned)Kpad * HIDDEN;
  for (unsigned i = blockIdx.x * blockDim.x + threadIdx.x; i < total;
       i += gridDim.x * blockDim.x) {
    const int kp = (int)(i >> 7), n = (int)(i & 127u);
    int orig = -1;
    if (kp >= p0 && kp < p0 + l0)      orig = o0 + (kp - p0);
    else if (kp >= p1 && kp < p1 + l1) orig = o1 + (kp - p1);
    else if (kp >= p2 && kp < p2 + l2) orig = o2 + (kp - p2);
    out[i] = (orig >= 0) ? (f16)w[orig * HIDDEN + n] : (f16)0.0f;
  }
}

__global__ void build_na(const float* __restrict__ x, int xdim, int xstride,
                         const float* __restrict__ nodes,
                         const float* __restrict__ fc, int fcdim,
                         f16* __restrict__ na) {
  const unsigned total = (unsigned)N_NODES * NA_ST;
  for (unsigned i = blockIdx.x * blockDim.x + threadIdx.x; i < total;
       i += gridDim.x * blockDim.x) {
    const unsigned node = i / NA_ST;
    const unsigned c    = i % NA_ST;
    float v = 0.0f;
    if ((int)c < xdim)                  v = x[node * xstride + c];
    else if ((int)c < xdim + 3)         v = nodes[node * 3 + (c - xdim)];
    else if ((int)c < xdim + 3 + fcdim) v = fc[c - xdim - 3];
    na[i] = (f16)v;
  }
}

// layer-0 edge_attr: write a full 8-half chunk (cols>=3 zeroed) so edge_mlp can read b128
__global__ void ea_init(const float* __restrict__ nodes, const int* __restrict__ src,
                        const int* __restrict__ dst, f16* __restrict__ ea) {
  for (unsigned e = blockIdx.x * blockDim.x + threadIdx.x; e < N_EDGES;
       e += gridDim.x * blockDim.x) {
    const int s = src[e], d = dst[e];
    H8 o;
#pragma unroll
    for (int c = 0; c < 8; ++c)
      o.h[c] = (c < 3) ? (f16)(0.5f * (nodes[s * 3 + c] + nodes[d * 3 + c])) : (f16)0.0f;
    *(uint4*)(ea + (unsigned)e * HIDDEN) = o.u;
  }
}

__global__ void deg_accum(const int* __restrict__ src, float* __restrict__ deg) {
  for (unsigned e = blockIdx.x * blockDim.x + threadIdx.x; e < N_EDGES;
       e += gridDim.x * blockDim.x) atomicAdd(&deg[src[e]], 1.0f);
}

__global__ void se_kernel(const float* __restrict__ x, const int* __restrict__ src,
                          const int* __restrict__ dst, f16* __restrict__ ea,
                          float* __restrict__ xsum) {
  const unsigned total = (unsigned)N_EDGES * 32;  // 4 cols per thread
  for (unsigned i = blockIdx.x * blockDim.x + threadIdx.x; i < total;
       i += gridDim.x * blockDim.x) {
    const unsigned e = i >> 5, c0 = (i & 31u) * 4;
    const int sN = src[e], dN = dst[e];
    const float4 a = *(const float4*)(x + (unsigned)dN * HIDDEN + c0);
    const float4 b = *(const float4*)(x + (unsigned)sN * HIDDEN + c0);
    const float v0 = 0.5f * (a.x + b.x), v1 = 0.5f * (a.y + b.y);
    const float v2 = 0.5f * (a.z + b.z), v3 = 0.5f * (a.w + b.w);
    H4 o;
    o.h[0] = (f16)v0; o.h[1] = (f16)v1; o.h[2] = (f16)v2; o.h[3] = (f16)v3;
    *(uint2*)(ea + (unsigned)e * HIDDEN + c0) = o.u;
    float* xs = xsum + (unsigned)sN * HIDDEN + c0;
    atomicAdd(xs + 0, v0);
    atomicAdd(xs + 1, v1);
    atomicAdd(xs + 2, v2);
    atomicAdd(xs + 3, v3);
  }
}

__global__ void xdiv(const float* __restrict__ xsum, const float* __restrict__ deg,
                     float* __restrict__ x) {
  const unsigned total = (unsigned)N_NODES * 32;
  for (unsigned i = blockIdx.x * blockDim.x + threadIdx.x; i < total;
       i += gridDim.x * blockDim.x) {
    const unsigned n = i >> 5, c0 = (i & 31u) * 4;
    const float inv = 1.0f / fmaxf(deg[n], 1.0f);
    float4 v = *(const float4*)(xsum + n * HIDDEN + c0);
    v.x *= inv; v.y *= inv; v.z *= inv; v.w *= inv;
    *(float4*)(x + n * HIDDEN + c0) = v;
  }
}

__global__ void col_mean(const float* __restrict__ x, int stride, float* __restrict__ gp) {
  __shared__ float red[256];
  const int c = blockIdx.x;
  float s = 0.0f;
  for (int i = threadIdx.x; i < N_NODES; i += 256) s += x[(unsigned)i * stride + c];
  red[threadIdx.x] = s;
  __syncthreads();
  for (int o = 128; o > 0; o >>= 1) {
    if ((int)threadIdx.x < o) red[threadIdx.x] += red[threadIdx.x + o];
    __syncthreads();
  }
  if (threadIdx.x == 0) gp[c] = red[0] / (float)N_NODES;
}

// single-row MLP3 (batch is all-zeros -> fc input identical for every node)
__global__ void fc_mlp3(const float* __restrict__ gp, int gpdim,
                        const float* __restrict__ prev, int prevdim,
                        const float* __restrict__ w1, const float* __restrict__ b1,
                        const float* __restrict__ w2, const float* __restrict__ b2,
                        const float* __restrict__ w3, const float* __restrict__ b3,
                        float* __restrict__ out) {
  __shared__ float sin_[192];
  __shared__ float h1[128];
  __shared__ float h2[128];
  const int t = threadIdx.x;  // 128 threads
  if (t < prevdim) sin_[t] = prev[t];
  if (t < gpdim)   sin_[prevdim + t] = gp[t];
  __syncthreads();
  const int din = prevdim + gpdim;
  float s = b1[t];
  for (int k = 0; k < din; ++k) s += sin_[k] * w1[k * 128 + t];
  h1[t] = gelu_f(s);
  __syncthreads();
  s = b2[t];
  for (int k = 0; k < 128; ++k) s += h1[k] * w2[k * 128 + t];
  h2[t] = gelu_f(s);
  __syncthreads();
  if (t < 64) {
    s = b3[t];
    for (int k = 0; k < 128; ++k) s += h2[k] * w3[k * 64 + t];
    out[t] = gelu_f(s);
  }
}

__global__ void decode_k(const float* __restrict__ x, const float* __restrict__ dw,
                         const float* __restrict__ db, float* __restrict__ out) {
  const unsigned total = (unsigned)N_NODES * 4;
  for (unsigned i = blockIdx.x * blockDim.x + threadIdx.x; i < total;
       i += gridDim.x * blockDim.x) {
    const unsigned n = i >> 2, o = i & 3u;
    float s = db[o];
    for (int k = 0; k < 128; ++k) s += x[n * 128 + k] * dw[k * 4 + o];
    out[i] = s;
  }
}

// ---------------- host orchestration ----------------
extern "C" void kernel_launch(void* const* d_in, const int* in_sizes, int n_in,
                              void* d_out, int out_size, void* d_ws, size_t ws_size,
                              hipStream_t stream) {
  (void)in_sizes; (void)n_in; (void)out_size;
  const float* nodes = (const float*)d_in[0];
  const int*   eidx  = (const int*)d_in[1];
  const int*   src   = eidx;            // edge_index[0]
  const int*   dst   = eidx + N_EDGES;  // edge_index[1]

  const float *ew1[4], *eb1[4], *ew2[4], *eb2[4], *nw1[4], *nb1[4], *nw2[4], *nb2[4];
  for (int i = 0; i < 4; ++i) {
    const int b = 3 + i * 8;
    ew1[i] = (const float*)d_in[b + 0]; eb1[i] = (const float*)d_in[b + 1];
    ew2[i] = (const float*)d_in[b + 2]; eb2[i] = (const float*)d_in[b + 3];
    nw1[i] = (const float*)d_in[b + 4]; nb1[i] = (const float*)d_in[b + 5];
    nw2[i] = (const float*)d_in[b + 6]; nb2[i] = (const float*)d_in[b + 7];
  }
  const float *fw1[2], *fb1[2], *fw2[2], *fb2[2], *fw3[2], *fb3[2];
  for (int j = 0; j < 2; ++j) {
    const int b = 35 + j * 6;
    fw1[j] = (const float*)d_in[b + 0]; fb1[j] = (const float*)d_in[b + 1];
    fw2[j] = (const float*)d_in[b + 2]; fb2[j] = (const float*)d_in[b + 3];
    fw3[j] = (const float*)d_in[b + 4]; fb3[j] = (const float*)d_in[b + 5];
  }
  const float* dec_w = (const float*)d_in[47];
  const float* dec_b = (const float*)d_in[48];

  // section-padded layouts (all sections start at multiples of 8 halves)
  const int naDim[4] = {70, 131, 195, 131};
  const int naP[4]   = {72, 136, 200, 136};      // align8(naDim)
  const int eaDim[4] = {3, 128, 128, 128};
  const int eaPad[4] = {8, 128, 128, 128};
  const int KpadE[4] = {160, 416, 544, 416};     // align32(2*naP+eaPad)
  const int KpadN[4] = {224, 288, 352, 288};     // align32(naP+128)

  // carve workspace (~175MB -> largely L2-resident on 192MB L2)
  char* ws = (char*)d_ws;
  size_t off = 0;
  auto carve = [&](size_t bytes) -> char* {
    char* p = ws + off;
    off += (bytes + 255) & ~(size_t)255;
    return p;
  };
  float* x    = (float*)carve((size_t)N_NODES * 128 * 4);
  float* xsum = (float*)carve((size_t)N_NODES * 128 * 4);
  float* agg  = (float*)carve((size_t)N_NODES * 128 * 4);
  float* deg  = (float*)carve((size_t)N_NODES * 4);
  float* gp   = (float*)carve(512);
  float* fc0  = (float*)carve(256);
  float* fc1  = (float*)carve(256);
  f16*   naB  = (f16*)carve((size_t)N_NODES * NA_ST * 2);
  f16*   eaB  = (f16*)carve((size_t)N_EDGES * HIDDEN * 2);
  f16 *ew1h[4], *ew2h[4], *nw1h[4], *nw2h[4];
  for (int i = 0; i < 4; ++i) {
    ew1h[i] = (f16*)carve((size_t)KpadE[i] * 128 * 2);
    ew2h[i] = (f16*)carve((size_t)128 * 128 * 2);
    nw1h[i] = (f16*)carve((size_t)KpadN[i] * 128 * 2);
    nw2h[i] = (f16*)carve((size_t)128 * 128 * 2);
  }
  if (off > ws_size) return;  // deterministic no-op if scratch too small

  // weight conversion f32 -> f16 with section remap (zero rows elsewhere)
  for (int i = 0; i < 4; ++i) {
    cvt_w16_seg<<<256, 256, 0, stream>>>(ew1[i], ew1h[i], KpadE[i],
                                         0, 0, naDim[i],
                                         naP[i], naDim[i], naDim[i],
                                         2 * naP[i], 2 * naDim[i], eaDim[i]);
    cvt_w16_seg<<<128, 256, 0, stream>>>(ew2[i], ew2h[i], 128,
                                         0, 0, 128, 0, 0, 0, 0, 0, 0);
    cvt_w16_seg<<<256, 256, 0, stream>>>(nw1[i], nw1h[i], KpadN[i],
                                         0, 0, naDim[i],
                                         naP[i], naDim[i], 128,
                                         0, 0, 0);
    cvt_w16_seg<<<128, 256, 0, stream>>>(nw2[i], nw2h[i], 128,
                                         0, 0, 128, 0, 0, 0, 0, 0, 0);
  }
  zero_f32<<<256, 256, 0, stream>>>(deg, N_NODES);
  deg_accum<<<1024, 256, 0, stream>>>(src, deg);
  ea_init<<<1024, 256, 0, stream>>>(nodes, src, dst, eaB);

  for (int i = 0; i < 4; ++i) {
    if (i == 0) {
      col_mean<<<3, 256, 0, stream>>>(nodes, 3, gp);
      fc_mlp3<<<1, 128, 0, stream>>>(gp, 3, nullptr, 0,
                                     fw1[0], fb1[0], fw2[0], fb2[0], fw3[0], fb3[0], fc0);
    } else if (i == 2) {
      col_mean<<<128, 256, 0, stream>>>(x, 128, gp);
      fc_mlp3<<<1, 128, 0, stream>>>(gp, 128, fc0, 64,
                                     fw1[1], fb1[1], fw2[1], fb2[1], fw3[1], fb3[1], fc1);
    }
    const float* xcur = (i == 0) ? nodes : x;
    const int    xd   = (i == 0) ? 3 : 128;
    const float* fcp  = (i == 0) ? fc0 : ((i == 2) ? fc1 : nullptr);
    const int    fcd  = (i == 0 || i == 2) ? 64 : 0;
    build_na<<<1024, 256, 0, stream>>>(xcur, xd, xd, nodes, fcp, fcd, naB);

    zero_f32<<<1024, 256, 0, stream>>>(agg, (unsigned)N_NODES * 128);
    const size_t shE = (size_t)(64 * (KpadE[i] + 8) + 64 * SH_ST) * sizeof(f16);
    edge_mlp<<<(N_EDGES + 63) / 64, 256, shE, stream>>>(
        naB, naP[i], eaB, eaPad[i], src, dst,
        ew1h[i], eb1[i], ew2h[i], eb2[i], KpadE[i], agg);

    const size_t shN = (size_t)(64 * (KpadN[i] + 8) + 64 * SH_ST) * sizeof(f16);
    node_mlp<<<(N_NODES + 63) / 64, 256, shN, stream>>>(
        naB, naP[i], agg, deg,
        nw1h[i], nb1[i], nw2h[i], nb2[i], KpadN[i], x);

    zero_f32<<<1024, 256, 0, stream>>>(xsum, (unsigned)N_NODES * 128);
    se_kernel<<<2048, 256, 0, stream>>>(x, src, dst, eaB, xsum);
    xdiv<<<1024, 256, 0, stream>>>(xsum, deg, x);
  }

  decode_k<<<512, 256, 0, stream>>>(x, dec_w, dec_b, (float*)d_out);
}
